// OptimizedEvolvableBlock_50895362457842
// MI455X (gfx1250) — matrix operand order repaired and checked by reading, not compiled, not run on hardware
//
#include <hip/hip_runtime.h>

typedef __attribute__((ext_vector_type(16))) _Float16 v16h;
typedef __attribute__((ext_vector_type(8)))  _Float16 v8h;
typedef __attribute__((ext_vector_type(8)))  float    v8f;

union AF { v16h v; v8h h2[2]; };
union CF { v8f v; float f[8]; };

#define CC 256       // out channels
#define PP 65536     // B*H*W = 64*32*32
#define EPSV 1e-5f

// ---------------------------------------------------------------------------
// sel: softmax -> top-3 -> threshold -> renormalize (single thread, trivial)
// ---------------------------------------------------------------------------
__global__ void sel_kernel(const float* __restrict__ aw, float* __restrict__ sel) {
    if (threadIdx.x != 0) return;
    float w[8]; float mx = -3.4e38f;
    for (int i = 0; i < 8; ++i) { w[i] = aw[i]; mx = fmaxf(mx, w[i]); }
    float s = 0.f;
    for (int i = 0; i < 8; ++i) { w[i] = expf(w[i] - mx); s += w[i]; }
    for (int i = 0; i < 8; ++i) w[i] /= s;
    float o[8]; bool taken[8];
    for (int i = 0; i < 8; ++i) { o[i] = 0.f; taken[i] = false; }
    for (int r = 0; r < 3; ++r) {
        int bi = 0; float bv = -1.f;
        for (int i = 0; i < 8; ++i)
            if (!taken[i] && w[i] > bv) { bv = w[i]; bi = i; }
        taken[bi] = true; o[bi] = bv;
    }
    float tot = 0.f;
    for (int i = 0; i < 8; ++i) { if (o[i] < 0.01f) o[i] = 0.f; tot += o[i]; }
    tot = fmaxf(tot, 1e-12f);
    for (int i = 0; i < 8; ++i) sel[i] = o[i] / tot;
}

// ---------------------------------------------------------------------------
// Implicit-GEMM conv via v_wmma_f32_16x16x32_f16.
//   D[co,p] = sum_k A[co,k] * B[k,p],  k = ci*KS*KS + tap, p = n*1024+y*32+x
// SRC: 0 = x, f32 NCHW (Cin=128); 1 = h, f32 [C][N*HW]; 2 = dw out, f16 [C][N*HW]
// Block: 256 thr (8 waves), tile M=128 x N=64, each wave does 2x2 WMMA tiles.
// ---------------------------------------------------------------------------
template<int KS, int DIL, int SRC>
__global__ __launch_bounds__(256)
void conv_gemm(const float* __restrict__ Ag, const void* __restrict__ Bg,
               float* __restrict__ raw, int Cin,
               const float* __restrict__ sel, int selIdx)
{
    if (selIdx >= 0 && sel[selIdx] == 0.0f) return;   // uniform scalar skip
    constexpr int T = KS * KS;
    constexpr int PAD = (KS / 2) * DIL;
    const int K   = Cin * T;
    const int co0 = blockIdx.y * 128;
    const int p0  = blockIdx.x * 64;
    const int t    = threadIdx.x;
    const int lane = t & 31;
    const int wave = t >> 5;
    const int wm = wave >> 1;          // 0..3 (M sub-tile group)
    const int wn = wave & 1;           // 0..1 (N sub-tile group)
    const int m  = lane & 15;
    const int hi = lane >> 4;          // lane half (ISA fragment layout)

    __shared__ __align__(32) _Float16 lA[128 * 32];   // [row][kk]
    __shared__ __align__(32) _Float16 lB[64 * 32];    // [col][kk]  (frag-order)

    CF acc[2][2];
    #pragma unroll
    for (int i = 0; i < 2; ++i)
        #pragma unroll
        for (int j = 0; j < 2; ++j)
            acc[i][j].v = (v8f){0.f,0.f,0.f,0.f,0.f,0.f,0.f,0.f};

    const int arow = t >> 3;           // 0..31
    const int akol = (t & 7) * 4;      // 0..28

    for (int k0 = 0; k0 < K; k0 += 32) {
        // ---- stage A: 128x32 weights, f32 -> f16 (coalesced float4 reads)
        #pragma unroll
        for (int rr = 0; rr < 4; ++rr) {
            const int row = arow + rr * 32;
            const float4 v = *(const float4*)(Ag + (size_t)(co0 + row) * K + k0 + akol);
            _Float16* d = &lA[row * 32 + akol];
            d[0] = (_Float16)v.x; d[1] = (_Float16)v.y;
            d[2] = (_Float16)v.z; d[3] = (_Float16)v.w;
        }
        if (k0 + 32 < K)
            __builtin_prefetch(Ag + (size_t)(co0 + arow) * K + k0 + 32, 0, 1);

        // ---- stage B: im2col gather, 32(K) x 64(cols), frag-order in LDS
        #pragma unroll
        for (int u = 0; u < 8; ++u) {
            const int idx = t * 8 + u;
            const int kk  = idx >> 6;
            const int col = idx & 63;
            const int k   = k0 + kk;
            int ci, dy, dx;
            if (KS == 1) { ci = k; dy = 0; dx = 0; }
            else {
                ci = k / T;
                const int tt = k - ci * T;
                dy = (tt / KS) * DIL - PAD;
                dx = (tt % KS) * DIL - PAD;
            }
            const int p  = p0 + col;
            const int n  = p >> 10;
            const int yy = ((p >> 5) & 31) + dy;
            const int xx = (p & 31) + dx;
            _Float16 v = (_Float16)0.f;
            if ((unsigned)yy < 32u && (unsigned)xx < 32u) {
                if (SRC == 0)
                    v = (_Float16)((const float*)Bg)[((size_t)n * 128 + ci) * 1024 + yy * 32 + xx];
                else if (SRC == 1)
                    v = (_Float16)((const float*)Bg)[((size_t)ci << 16) + (n << 10) + yy * 32 + xx];
                else
                    v = ((const _Float16*)Bg)[((size_t)ci << 16) + (n << 10) + yy * 32 + xx];
            }
            lB[col * 32 + kk] = v;
        }
        __syncthreads();

        // ---- load fragments (vectorized LDS reads, contiguous k = hi*16 + s in A and B)
        AF af[2]; v16h bf[2];
        #pragma unroll
        for (int mi = 0; mi < 2; ++mi) {
            const _Float16* ap = &lA[(wm * 32 + mi * 16 + m) * 32 + hi * 16];
            af[mi].h2[0] = *(const v8h*)ap;          // K = hi*16 + 0..7
            af[mi].h2[1] = *(const v8h*)(ap + 8);    // K = hi*16 + 8..15
        }
        #pragma unroll
        for (int ni = 0; ni < 2; ++ni) {
            const _Float16* bp = &lB[(wn * 32 + ni * 16 + m) * 32 + hi * 16];
            bf[ni] = *(const v16h*)bp;               // K = hi*16 + 0..15
        }
        #pragma unroll
        for (int mi = 0; mi < 2; ++mi)
            #pragma unroll
            for (int ni = 0; ni < 2; ++ni)
                acc[mi][ni].v = __builtin_amdgcn_wmma_f32_16x16x32_f16(
                    false, af[mi].v, false, bf[ni],
                    (short)0, acc[mi][ni].v, false, false);
        __syncthreads();
    }

    // ---- epilogue: C/D layout -> raw[co][p]
    #pragma unroll
    for (int mi = 0; mi < 2; ++mi)
        #pragma unroll
        for (int ni = 0; ni < 2; ++ni) {
            const int colg = p0 + wn * 32 + ni * 16 + m;
            #pragma unroll
            for (int v = 0; v < 8; ++v) {
                const int rowg = co0 + wm * 32 + mi * 16 + hi * 8 + v;
                raw[((size_t)rowg << 16) + colg] = acc[mi][ni].f[v];
            }
        }
}

// ---------------------------------------------------------------------------
// Per-channel sum / sumsq over [C][P] (one block per channel)
// ---------------------------------------------------------------------------
__global__ __launch_bounds__(256)
void channel_stats(const float* __restrict__ buf, float* __restrict__ st,
                   const float* __restrict__ sel, int selIdx)
{
    if (selIdx >= 0 && sel[selIdx] == 0.0f) return;
    const int c = blockIdx.x;
    const int t = threadIdx.x;
    float s = 0.f, ss = 0.f;
    for (int p = t; p < PP; p += 256) {
        const float v = buf[((size_t)c << 16) + p];
        s += v; ss += v * v;
    }
    __shared__ float r1[256], r2[256];
    r1[t] = s; r2[t] = ss; __syncthreads();
    for (int o = 128; o > 0; o >>= 1) {
        if (t < o) { r1[t] += r1[t + o]; r2[t] += r2[t + o]; }
        __syncthreads();
    }
    if (t == 0) { st[c] = r1[0]; st[CC + c] = r2[0]; }
}

__device__ __forceinline__ float bn_apply(float x, float sum, float sumsq,
                                          float g, float b) {
    const float mean = sum * (1.f / (float)PP);
    const float var  = sumsq * (1.f / (float)PP) - mean * mean;
    return (x - mean) * rsqrtf(var + EPSV) * g + b;
}

// ---------------------------------------------------------------------------
// h = relu(BN(raw))  stored channel-major [C][N*HW]
// ---------------------------------------------------------------------------
__global__ __launch_bounds__(256)
void bn_relu_h(const float* __restrict__ raw, const float* __restrict__ st,
               const float* __restrict__ g, const float* __restrict__ b,
               float* __restrict__ h)
{
    const int i = blockIdx.x * 256 + threadIdx.x;     // [C][P]
    const int c = i >> 16;
    const float v = bn_apply(raw[i], st[c], st[CC + c], g[c], b[c]);
    h[i] = v > 0.f ? v : 0.f;
}

__device__ __forceinline__ void pool3(const float* __restrict__ hc, int n,
                                      int y, int x, float& mx, float& av) {
    const float* hn = hc + (n << 10);
    mx = -3.4e38f; float s = 0.f;
    #pragma unroll
    for (int dy = -1; dy <= 1; ++dy) {
        const int yy = y + dy;
        if ((unsigned)yy >= 32u) continue;
        #pragma unroll
        for (int dx = -1; dx <= 1; ++dx) {
            const int xx = x + dx;
            if ((unsigned)xx >= 32u) continue;
            const float v = hn[yy * 32 + xx];
            mx = fmaxf(mx, v);
            s += v;
        }
    }
    av = s * (1.f / 9.f);     // count_include_pad
}

// ---------------------------------------------------------------------------
// channel stats of maxpool3(h) and avgpool3(h) (recomputed on the fly)
// ---------------------------------------------------------------------------
__global__ __launch_bounds__(256)
void pool_stats_k(const float* __restrict__ h, float* __restrict__ stm,
                  float* __restrict__ sta)
{
    const int c = blockIdx.x;
    const int t = threadIdx.x;
    const float* hc = h + ((size_t)c << 16);
    float sm = 0.f, ssm = 0.f, sa = 0.f, ssa = 0.f;
    for (int p = t; p < PP; p += 256) {
        float mx, av;
        pool3(hc, p >> 10, (p >> 5) & 31, p & 31, mx, av);
        sm += mx; ssm += mx * mx; sa += av; ssa += av * av;
    }
    __shared__ float red[256];
    float vals[4] = { sm, ssm, sa, ssa };
    float outv[4];
    for (int q = 0; q < 4; ++q) {
        red[t] = vals[q]; __syncthreads();
        for (int o = 128; o > 0; o >>= 1) {
            if (t < o) red[t] += red[t + o];
            __syncthreads();
        }
        outv[q] = red[0]; __syncthreads();
    }
    if (t == 0) { stm[c] = outv[0]; stm[CC + c] = outv[1];
                  sta[c] = outv[2]; sta[CC + c] = outv[3]; }
}

// ---------------------------------------------------------------------------
// out = sel1*BN(maxpool(h)) + sel2*BN(avgpool(h)) + sel3*h   (initializes out)
// ---------------------------------------------------------------------------
__global__ __launch_bounds__(256)
void pool_skip_k(const float* __restrict__ h, const float* __restrict__ stm,
                 const float* __restrict__ sta,
                 const float* __restrict__ gmp, const float* __restrict__ bmp,
                 const float* __restrict__ gap, const float* __restrict__ bap,
                 const float* __restrict__ sel, float* __restrict__ out)
{
    const int i = blockIdx.x * 256 + threadIdx.x;     // [C][P]
    const int c = i >> 16;
    const int p = i & (PP - 1);
    const int n = p >> 10;
    float mx, av;
    pool3(h + ((size_t)c << 16), n, (p >> 5) & 31, p & 31, mx, av);
    float v = sel[3] * h[i];
    v += sel[1] * bn_apply(mx, stm[c], stm[CC + c], gmp[c], bmp[c]);
    v += sel[2] * bn_apply(av, sta[c], sta[CC + c], gap[c], bap[c]);
    out[(((size_t)n * CC + c) << 10) + (p & 1023)] = v;   // NCHW
}

// ---------------------------------------------------------------------------
// out += sel[si] * BN(raw)
// ---------------------------------------------------------------------------
__global__ __launch_bounds__(256)
void bn_accum_k(const float* __restrict__ raw, const float* __restrict__ st,
                const float* __restrict__ g, const float* __restrict__ b,
                const float* __restrict__ sel, int si, float* __restrict__ out)
{
    const float s = sel[si];
    if (s == 0.0f) return;
    const int i = blockIdx.x * 256 + threadIdx.x;     // [C][P]
    const int c = i >> 16;
    const int p = i & (PP - 1);
    const int n = p >> 10;
    out[(((size_t)n * CC + c) << 10) + (p & 1023)] +=
        s * bn_apply(raw[i], st[c], st[CC + c], g[c], b[c]);
}

// ---------------------------------------------------------------------------
// depthwise 3x3 (groups=C), pad 1 -> f16 [C][N*HW] for the pointwise GEMM
// ---------------------------------------------------------------------------
__global__ __launch_bounds__(256)
void dw_conv_k(const float* __restrict__ h, const float* __restrict__ wdw,
               _Float16* __restrict__ dwo, const float* __restrict__ sel, int si)
{
    if (sel[si] == 0.0f) return;
    const int i = blockIdx.x * 256 + threadIdx.x;     // [C][P]
    const int c = i >> 16;
    const int p = i & (PP - 1);
    const int y = (p >> 5) & 31, x = p & 31;
    const float* hn = h + ((size_t)c << 16) + ((p >> 10) << 10);
    const float* wc = wdw + c * 9;
    float acc = 0.f;
    #pragma unroll
    for (int dy = 0; dy < 3; ++dy) {
        const int yy = y + dy - 1;
        if ((unsigned)yy >= 32u) continue;
        #pragma unroll
        for (int dx = 0; dx < 3; ++dx) {
            const int xx = x + dx - 1;
            if ((unsigned)xx >= 32u) continue;
            acc += wc[dy * 3 + dx] * hn[yy * 32 + xx];
        }
    }
    dwo[i] = (_Float16)acc;
}

// ---------------------------------------------------------------------------
extern "C" void kernel_launch(void* const* d_in, const int* in_sizes, int n_in,
                              void* d_out, int out_size, void* d_ws, size_t ws_size,
                              hipStream_t stream)
{
    const float* x    = (const float*)d_in[0];
    const float* aw   = (const float*)d_in[1];
    const float* w_in = (const float*)d_in[2];
    const float* g_in = (const float*)d_in[3];
    const float* b_in = (const float*)d_in[4];
    const float* w3   = (const float*)d_in[5];
    const float* g3   = (const float*)d_in[6];
    const float* b3   = (const float*)d_in[7];
    const float* w5   = (const float*)d_in[8];
    const float* g5   = (const float*)d_in[9];
    const float* b5   = (const float*)d_in[10];
    const float* wdw  = (const float*)d_in[11];
    const float* wpw  = (const float*)d_in[12];
    const float* gs   = (const float*)d_in[13];
    const float* bs   = (const float*)d_in[14];
    const float* wd   = (const float*)d_in[15];
    const float* gd   = (const float*)d_in[16];
    const float* bd   = (const float*)d_in[17];
    const float* gmp  = (const float*)d_in[18];
    const float* bmp  = (const float*)d_in[19];
    const float* gap  = (const float*)d_in[20];
    const float* bap  = (const float*)d_in[21];
    float* out = (float*)d_out;

    char* ws = (char*)d_ws;
    float*    h   = (float*)(ws);                                   // 64 MB
    float*    raw = (float*)(ws + (size_t)64  * 1024 * 1024);       // 64 MB
    _Float16* dwb = (_Float16*)(ws + (size_t)128 * 1024 * 1024);    // 32 MB
    float*    st  = (float*)(ws + (size_t)160 * 1024 * 1024);
    float* sel  = st;
    float* stin = st + 8;
    float* stop = st + 8 + 512;
    float* stmp = st + 8 + 1024;
    float* stap = st + 8 + 1536;

    const dim3 gg(PP / 64, 2);
    const int NB = (CC * PP) / 256;   // 65536 blocks for elementwise kernels

    sel_kernel<<<1, 32, 0, stream>>>(aw, sel);

    // input transition: 1x1 conv (WMMA GEMM, Cin=128) -> BN -> ReLU -> h
    conv_gemm<1,1,0><<<gg, 256, 0, stream>>>(w_in, x, raw, 128, sel, -1);
    channel_stats<<<CC, 256, 0, stream>>>(raw, stin, sel, -1);
    bn_relu_h<<<NB, 256, 0, stream>>>(raw, stin, g_in, b_in, h);

    // pool + skip branches (initialize out)
    pool_stats_k<<<CC, 256, 0, stream>>>(h, stmp, stap);
    pool_skip_k<<<NB, 256, 0, stream>>>(h, stmp, stap, gmp, bmp, gap, bap, sel, out);

    // conv 3x3
    conv_gemm<3,1,1><<<gg, 256, 0, stream>>>(w3, h, raw, 256, sel, 4);
    channel_stats<<<CC, 256, 0, stream>>>(raw, stop, sel, 4);
    bn_accum_k<<<NB, 256, 0, stream>>>(raw, stop, g3, b3, sel, 4, out);

    // conv 5x5
    conv_gemm<5,1,1><<<gg, 256, 0, stream>>>(w5, h, raw, 256, sel, 5);
    channel_stats<<<CC, 256, 0, stream>>>(raw, stop, sel, 5);
    bn_accum_k<<<NB, 256, 0, stream>>>(raw, stop, g5, b5, sel, 5, out);

    // dilated conv 3x3 (dil=2)
    conv_gemm<3,2,1><<<gg, 256, 0, stream>>>(wd, h, raw, 256, sel, 7);
    channel_stats<<<CC, 256, 0, stream>>>(raw, stop, sel, 7);
    bn_accum_k<<<NB, 256, 0, stream>>>(raw, stop, gd, bd, sel, 7, out);

    // separable: depthwise 3x3 -> pointwise 1x1 (WMMA GEMM) -> BN
    dw_conv_k<<<NB, 256, 0, stream>>>(h, wdw, dwb, sel, 6);
    conv_gemm<1,1,2><<<gg, 256, 0, stream>>>(wpw, dwb, raw, 256, sel, 6);
    channel_stats<<<CC, 256, 0, stream>>>(raw, stop, sel, 6);
    bn_accum_k<<<NB, 256, 0, stream>>>(raw, stop, gs, bs, sel, 6, out);
}
